// RNNModel_54065048322711
// MI455X (gfx1250) — matrix operand order repaired and checked by reading, not compiled, not run on hardware
//
#include <hip/hip_runtime.h>
#include <hip/hip_bf16.h>

typedef __bf16 bf16;
typedef __attribute__((ext_vector_type(16))) __bf16 v16bf;
typedef __attribute__((ext_vector_type(8)))  __bf16 v8bf;
typedef __attribute__((ext_vector_type(8)))  float  v8f;
typedef __attribute__((ext_vector_type(4)))  float  v4f;

// Model dims
#define BB   256
#define TT   512
#define HH   256
#define FIN  128
#define G3H  768   // 3*H
#define NOUT 128   // W*COUT
#define MROWS (BB*TT)  // 131072

__device__ inline v8f zero8() {
    v8f z = {0.f,0.f,0.f,0.f,0.f,0.f,0.f,0.f};
    return z;
}

// ---- Fragment loaders -------------------------------------------------------
// A-matrix 16x32 bf16 layout (ISA 7.12.2): lane L: m=L&15, kh=L>>4;
// holds K = kh*8+{0..7} (v0..v3) and 16+kh*8+{0..7} (v4..v7).
__device__ inline v16bf load_a_f32(const float* __restrict__ A, int lda,
                                   int row0, int kbase, int lane) {
    int m = lane & 15, kh = lane >> 4;
    const float* p = A + (size_t)(row0 + m) * lda + kbase + kh * 8;
    v4f f0 = *(const v4f*)(p);
    v4f f1 = *(const v4f*)(p + 4);
    v4f f2 = *(const v4f*)(p + 16);
    v4f f3 = *(const v4f*)(p + 20);
    v16bf a;
#pragma unroll
    for (int i = 0; i < 4; ++i) {
        a[i]      = (bf16)f0[i];
        a[4 + i]  = (bf16)f1[i];
        a[8 + i]  = (bf16)f2[i];
        a[12 + i] = (bf16)f3[i];
    }
    return a;
}

__device__ inline v16bf load_a_bf16(const bf16* __restrict__ A, int lda,
                                    int row0, int kbase, int lane) {
    int m = lane & 15, kh = lane >> 4;
    const bf16* p = A + (size_t)(row0 + m) * lda + kbase + kh * 8;
    v8bf x0 = *(const v8bf*)(p);
    v8bf x1 = *(const v8bf*)(p + 16);
    v16bf a;
#pragma unroll
    for (int i = 0; i < 8; ++i) { a[i] = x0[i]; a[8 + i] = x1[i]; }
    return a;
}

// B-matrix 32x16 bf16: lane L: n=L&15, K = (L>>4)*8+{0..7} and 16+(L>>4)*8+{0..7}
// (matches A). Weights stored pre-transposed as [N][K] bf16.
__device__ inline v16bf load_b(const bf16* __restrict__ Wt, int ldk,
                               int n0, int kbase, int lane) {
    int n = n0 + (lane & 15), kh = lane >> 4;
    const bf16* p = Wt + (size_t)n * ldk + kbase + kh * 8;
    v8bf x0 = *(const v8bf*)(p);
    v8bf x1 = *(const v8bf*)(p + 16);
    v16bf b;
#pragma unroll
    for (int i = 0; i < 8; ++i) { b[i] = x0[i]; b[8 + i] = x1[i]; }
    return b;
}

__device__ inline float sigm(float x) { return 1.0f / (1.0f + __expf(-x)); }

// ---- Weight convert/transpose/fold kernel ----------------------------------
// W1T [256][128], WkT [768][256], WrT [768][256], W2sT [128][256] (folded concat)
__global__ void convert_kernel(const float* __restrict__ W1, const float* __restrict__ Wk,
                               const float* __restrict__ Wr, const float* __restrict__ W2,
                               bf16* __restrict__ W1T, bf16* __restrict__ WkT,
                               bf16* __restrict__ WrT, bf16* __restrict__ W2sT) {
    int i = blockIdx.x * blockDim.x + threadIdx.x;
    if (i < 256 * 128) {                 // W1: (128,256) -> [n][k]
        int n = i >> 7, k = i & 127;
        W1T[i] = (bf16)W1[k * 256 + n];
        return;
    }
    i -= 256 * 128;
    if (i < 768 * 256) {                 // Wk: (256,768) -> [n][k]
        int n = i >> 8, k = i & 255;
        WkT[i] = (bf16)Wk[k * 768 + n];
        return;
    }
    i -= 768 * 256;
    if (i < 768 * 256) {                 // Wr: (256,768) -> [n][k]
        int n = i >> 8, k = i & 255;
        WrT[i] = (bf16)Wr[k * 768 + n];
        return;
    }
    i -= 768 * 256;
    if (i < 128 * 256) {                 // W2 folded: (512,128) -> [n][k], k<256
        int n = i >> 8, k = i & 255;
        W2sT[i] = (bf16)(W2[k * 128 + n] + W2[(k + 256) * 128 + n]);
    }
}

// ---- Generic wave-tile GEMM: Out[M][N] = act(A[M][K] @ Wt^T + bias) --------
// Each wave owns a 16(M) x 64(N) tile; K-loop in steps of 32 with bf16 WMMA.
template <int K, int N, bool A_F32, bool RELU>
__global__ void gemm_bf16_kernel(const void* __restrict__ Aptr,
                                 const bf16* __restrict__ Wt,    // [N][K]
                                 const float* __restrict__ bias, // [N]
                                 bf16* __restrict__ Out,         // [M][N]
                                 int M) {
    int flat = blockIdx.x * blockDim.x + threadIdx.x;
    int wid  = flat >> 5;
    int lane = threadIdx.x & 31;
    const int NW = N / 64;
    int mt = wid / NW;
    int n0 = (wid % NW) * 64;
    if (mt * 16 >= M) return;           // wave-uniform (grid sized exactly)
    int row0 = mt * 16;

    v8f acc[4];
#pragma unroll
    for (int j = 0; j < 4; ++j) acc[j] = zero8();

    for (int kb = 0; kb < K; kb += 32) {
        v16bf a;
        if constexpr (A_F32) a = load_a_f32((const float*)Aptr, K, row0, kb, lane);
        else                 a = load_a_bf16((const bf16*)Aptr, K, row0, kb, lane);
#pragma unroll
        for (int j = 0; j < 4; ++j) {
            v16bf b = load_b(Wt, K, n0 + j * 16, kb, lane);
            acc[j] = __builtin_amdgcn_wmma_f32_16x16x32_bf16(
                false, a, false, b, (short)0, acc[j], false, false);
        }
    }

    int nl = lane & 15, sel = lane >> 4;
#pragma unroll
    for (int j = 0; j < 4; ++j) {
        int n = n0 + j * 16 + nl;
        float bv = bias[n];
#pragma unroll
        for (int i = 0; i < 8; ++i) {
            int m = i + sel * 8;
            float v = acc[j][i] + bv;
            if (RELU) v = v > 0.f ? v : 0.f;
            Out[(size_t)(row0 + m) * N + n] = (bf16)v;
        }
    }
}

// ---- Persistent GRU kernel --------------------------------------------------
// 16 workgroups x 16 batch rows each; h slice lives in LDS across all 512 steps.
// Phase 1: rec = h @ Wr via WMMA. kk-outer loop: 6 accumulators (one per
//   N-tile) stay live; per K-step one A fragment is rebuilt from LDS while a
//   12-load batch prefetches the NEXT K-step's six B fragments, so each K-step
//   exposes a single L2 latency hidden behind 6 WMMAs + stores (instead of one
//   wait-zero per tile). The opaque zero offset defeats LICM (which would
//   otherwise hoist 384 VGPRs of loop-invariant fragments into scratch
//   spills) without capturing the pointer, keeping global_load addressing.
// Phase 2: gate math + h update (256 threads, 16 elems each); br read via LDS.
__global__ void gru_kernel(const bf16* __restrict__ xg,   // [B][T][768]
                           const bf16* __restrict__ WrT,  // [768][256]
                           const float* __restrict__ br,  // [768]
                           float* __restrict__ h_last) {  // [B][256]
    __shared__ float hsh[16][264];   // padded vs 256 to dodge bank conflicts
    __shared__ float rec[16][776];   // padded vs 768
    __shared__ float brs[768];

    int tid  = threadIdx.x;
    int lane = tid & 31;
    int wv   = tid >> 5;             // 0..7
    int b0   = blockIdx.x * 16;

    for (int i = tid; i < 16 * 256; i += 256) hsh[i >> 8][i & 255] = 0.f;
    for (int i = tid; i < 768; i += 256) brs[i] = br[i];
    __syncthreads();

    int em  = tid >> 4;              // gate-phase row  0..15
    int ec0 = (tid & 15) * 16;       // gate-phase col base (0..240)

    int am = lane & 15, akh = lane >> 4;   // A-fragment lane decomposition

    for (int t = 0; t < TT; ++t) {
        // Opaque zero offset: defeats loop-invariant hoisting of the weight
        // loads without capturing the pointer (keeps global addrspace).
        unsigned zoff = 0;
        asm volatile("" : "+s"(zoff));
        const bf16* Wrp = WrT + zoff;

        // ---- Phase 1: rec = h @ Wr (bf16 WMMA, f32 accum) ----
        v8f acc[6];
#pragma unroll
        for (int j = 0; j < 6; ++j) acc[j] = zero8();

        v16bf bcur[6], bnext[6];
#pragma unroll
        for (int j = 0; j < 6; ++j)          // kk=0 batch in flight first
            bcur[j] = load_b(Wrp, HH, (wv * 6 + j) * 16, 0, lane);

#pragma unroll
        for (int kk = 0; kk < 8; ++kk) {
            if (kk < 7) {                    // prefetch kk+1's six fragments
#pragma unroll
                for (int j = 0; j < 6; ++j)
                    bnext[j] = load_b(Wrp, HH, (wv * 6 + j) * 16,
                                      (kk + 1) * 32, lane);
            }
            // Rebuild the A fragment for this K-step from LDS (overlaps the
            // global-load batch; keeps VGPR pressure low).
            v16bf a;
            {
                const float* p = &hsh[am][kk * 32 + akh * 8];
                v4f f0 = *(const v4f*)(p);
                v4f f1 = *(const v4f*)(p + 4);
                v4f f2 = *(const v4f*)(p + 16);
                v4f f3 = *(const v4f*)(p + 20);
#pragma unroll
                for (int i = 0; i < 4; ++i) {
                    a[i]      = (bf16)f0[i];
                    a[4 + i]  = (bf16)f1[i];
                    a[8 + i]  = (bf16)f2[i];
                    a[12 + i] = (bf16)f3[i];
                }
            }
#pragma unroll
            for (int j = 0; j < 6; ++j)
                acc[j] = __builtin_amdgcn_wmma_f32_16x16x32_bf16(
                    false, a, false, bcur[j], (short)0, acc[j], false, false);
            if (kk < 7) {
#pragma unroll
                for (int j = 0; j < 6; ++j) bcur[j] = bnext[j];
            }
        }

        {
            int sel = lane >> 4, nl = lane & 15;
#pragma unroll
            for (int j = 0; j < 6; ++j) {
                int n = (wv * 6 + j) * 16 + nl;
#pragma unroll
                for (int i = 0; i < 8; ++i) rec[i + sel * 8][n] = acc[j][i];
            }
        }
        __syncthreads();

        // ---- Phase 2: gates + h update ----
        const bf16* xrow = xg + ((size_t)(b0 + em) * TT + t) * G3H;
#pragma unroll
        for (int j = 0; j < 16; ++j) {
            int n = ec0 + j;
            float xz = (float)xrow[n];
            float xr = (float)xrow[n + 256];
            float xh = (float)xrow[n + 512];
            float rz = rec[em][n]       + brs[n];
            float rr = rec[em][n + 256] + brs[n + 256];
            float rh = rec[em][n + 512] + brs[n + 512];
            float z  = sigm(xz + rz);
            float r  = sigm(xr + rr);
            float hh = xh + r * rh;
            hh = hh > 0.f ? hh : 0.f;
            float hp = hsh[em][n];
            hsh[em][n] = z * hp + (1.f - z) * hh;
        }
        if (t + 1 < TT) {  // warm L2/WGP$ for the next timestep's gate row
            __builtin_prefetch(xrow + G3H + ec0, 0, 0);
            __builtin_prefetch(xrow + G3H + 512 + ec0, 0, 0);
        }
        __syncthreads();
    }

    for (int i = tid; i < 16 * 256; i += 256) {
        int m = i >> 8, n = i & 255;
        h_last[(size_t)(b0 + m) * HH + n] = hsh[m][n];
    }
}

// ---- Output GEMM: y = h_last @ W2sum + b2  (256x256 @ 256x128) -------------
__global__ void out_kernel(const float* __restrict__ h_last,
                           const bf16* __restrict__ W2sT,  // [128][256]
                           const float* __restrict__ b2,
                           float* __restrict__ y) {        // [256][128]
    int lane = threadIdx.x & 31;
    int wv   = threadIdx.x >> 5;     // 0..7 -> N-tile
    int row0 = blockIdx.x * 16;
    int n0   = wv * 16;
    v8f acc = zero8();
    for (int kb = 0; kb < HH; kb += 32) {
        v16bf a = load_a_f32(h_last, HH, row0, kb, lane);
        v16bf b = load_b(W2sT, HH, n0, kb, lane);
        acc = __builtin_amdgcn_wmma_f32_16x16x32_bf16(
            false, a, false, b, (short)0, acc, false, false);
    }
    int n = n0 + (lane & 15), sel = lane >> 4;
    float bv = b2[n];
#pragma unroll
    for (int i = 0; i < 8; ++i) {
        int m = i + sel * 8;
        y[(size_t)(row0 + m) * NOUT + n] = acc[i] + bv;
    }
}

// ---- Host launcher ----------------------------------------------------------
extern "C" void kernel_launch(void* const* d_in, const int* in_sizes, int n_in,
                              void* d_out, int out_size, void* d_ws, size_t ws_size,
                              hipStream_t stream) {
    const float* x  = (const float*)d_in[0];
    const float* W1 = (const float*)d_in[1];
    const float* b1 = (const float*)d_in[2];
    const float* Wk = (const float*)d_in[3];
    const float* Wr = (const float*)d_in[4];
    const float* bi = (const float*)d_in[5];
    const float* br = (const float*)d_in[6];
    const float* W2 = (const float*)d_in[7];
    const float* b2 = (const float*)d_in[8];
    float* y = (float*)d_out;

    char* ws = (char*)d_ws;
    size_t off = 0;
    auto alloc = [&](size_t bytes) {
        char* p = ws + off;
        off += (bytes + 255) & ~(size_t)255;
        return p;
    };
    bf16*  W1T  = (bf16*)alloc((size_t)256 * 128 * 2);
    bf16*  WkT  = (bf16*)alloc((size_t)768 * 256 * 2);
    bf16*  WrT  = (bf16*)alloc((size_t)768 * 256 * 2);
    bf16*  W2sT = (bf16*)alloc((size_t)128 * 256 * 2);
    bf16*  h1   = (bf16*)alloc((size_t)MROWS * HH * 2);    //  67 MB
    bf16*  xgb  = (bf16*)alloc((size_t)MROWS * G3H * 2);   // 201 MB
    float* hl   = (float*)alloc((size_t)BB * HH * 4);

    // 1) weight transpose/convert/fold
    convert_kernel<<<(32768 + 196608 + 196608 + 32768) / 256, 256, 0, stream>>>(
        W1, Wk, Wr, W2, W1T, WkT, WrT, W2sT);

    // 2) h1 = relu(x @ W1 + b1)   [131072 x 128] @ [128 x 256]
    {
        int waves = (MROWS / 16) * (HH / 64);
        gemm_bf16_kernel<FIN, HH, true, true>
            <<<waves * 32 / 256, 256, 0, stream>>>(x, W1T, b1, h1, MROWS);
    }
    // 3) xg = h1 @ Wk + bi        [131072 x 256] @ [256 x 768]
    {
        int waves = (MROWS / 16) * (G3H / 64);
        gemm_bf16_kernel<HH, G3H, false, false>
            <<<waves * 32 / 256, 256, 0, stream>>>(h1, WkT, bi, xgb, MROWS);
    }
    // 4) GRU recurrence (batch-partitioned, persistent over T)
    gru_kernel<<<BB / 16, 256, 0, stream>>>(xgb, WrT, br, hl);

    // 5) y = h_last @ (W2[:256]+W2[256:]) + b2
    out_kernel<<<BB / 16, 256, 0, stream>>>(hl, W2sT, b2, y);
}
